// NF4LayernormLinear_11192684773387
// MI455X (gfx1250) — matrix operand order repaired and checked by reading, not compiled, not run on hardware
//
#include <hip/hip_runtime.h>

// ---------------------------------------------------------------------------
// NF4 LayerNorm + Linear for gfx1250 (MI455X, wave32, WMMA)
//   out0: linear_out [4096, 12288] f32
//   out1: ln_out     [4096, 4096]  f32
//
// Data-movement plan (23.3 TB/s HBM, 192 MB L2):
//   * dequant NF4 codes (201 MB int32, NT-loaded, read ONCE) -> Wh f16 (96 MB)
//   * LN writes f32 ln_out (NT) + f16 A copy (32 MB) for WMMA.
//   * A (32 MB) + Wh (96 MB) both fit the 192 MB L2 -> GEMM operands are
//     L2-resident; NT-store the 192 MB C stream so it does not evict them.
// ---------------------------------------------------------------------------

typedef __attribute__((ext_vector_type(16))) _Float16 v16h;
typedef __attribute__((ext_vector_type(8)))  _Float16 v8h;
typedef __attribute__((ext_vector_type(4)))  _Float16 v4h;
typedef __attribute__((ext_vector_type(8)))  float    v8f;
typedef __attribute__((ext_vector_type(4)))  float    v4f;
typedef __attribute__((ext_vector_type(4)))  int      v4i;

#define IN_SIZE   4096
#define OUT_SIZE  12288
#define M_ROWS    4096
#define N_BLOCKS  786432      // OUT_SIZE*IN_SIZE/64
#define LN_EPS    1e-5f

__constant__ float NF4_CB[16] = {
    -1.0f, -0.6961928009986877f, -0.5250730514526367f, -0.39491748809814453f,
    -0.28444138169288635f, -0.18477343022823334f, -0.09105003625154495f, 0.0f,
    0.07958029955625534f, 0.16093020141124725f, 0.24611230194568634f,
    0.33791524171829224f, 0.44070982933044434f, 0.5626170039176941f,
    0.7229568362236023f, 1.0f};

// WGP-scope prefetch (scope field 0 = WGP: pull into all cache levels).
// __builtin_prefetch only reaches SE/SYS scopes, so use inline asm.
#define PREFETCH_WGP_512(p) \
  asm volatile("global_prefetch_b8 %0, off offset:512" ::"v"(p))

// ---------------------------------------------------------------------------
// Kernel 1: LayerNorm. One 256-thread block per row (16 elems/thread, v4f).
// Single fused mean/meansq reduction. NT f32 store + f16 copy for WMMA A.
// ---------------------------------------------------------------------------
__global__ __launch_bounds__(256) void ln_kernel(const float* __restrict__ x,
                                                 const float* __restrict__ gamma,
                                                 const float* __restrict__ beta,
                                                 float* __restrict__ y,
                                                 _Float16* __restrict__ yh) {
  __shared__ float redm[256];
  __shared__ float redv[256];
  const int row = blockIdx.x;
  const int tid = threadIdx.x;
  const v4f* xr = (const v4f*)(x + (size_t)row * IN_SIZE);

  v4f v[4];
  float s = 0.f, ss = 0.f;
#pragma unroll
  for (int i = 0; i < 4; ++i) {
    v[i] = xr[tid + i * 256];
    s  += v[i].x + v[i].y + v[i].z + v[i].w;
    ss += v[i].x * v[i].x + v[i].y * v[i].y + v[i].z * v[i].z + v[i].w * v[i].w;
  }

  redm[tid] = s;
  redv[tid] = ss;
  __syncthreads();
  for (int off = 128; off > 0; off >>= 1) {
    if (tid < off) {
      redm[tid] += redm[tid + off];
      redv[tid] += redv[tid + off];
    }
    __syncthreads();
  }
  const float mu  = redm[0] * (1.f / IN_SIZE);
  const float var = redv[0] * (1.f / IN_SIZE) - mu * mu;
  const float rs  = __frsqrt_rn(var + LN_EPS);

  v4f* yr = (v4f*)(y + (size_t)row * IN_SIZE);
  _Float16* yhr = yh + (size_t)row * IN_SIZE;
#pragma unroll
  for (int i = 0; i < 4; ++i) {
    const int c = tid + i * 256;
    const v4f gv = ((const v4f*)gamma)[c];
    const v4f bv = ((const v4f*)beta)[c];
    v4f o;
    o.x = (v[i].x - mu) * rs * gv.x + bv.x;
    o.y = (v[i].y - mu) * rs * gv.y + bv.y;
    o.z = (v[i].z - mu) * rs * gv.z + bv.z;
    o.w = (v[i].w - mu) * rs * gv.w + bv.w;
    __builtin_nontemporal_store(o, yr + c);   // pure output stream
    v4h h;
    h[0] = (_Float16)o.x; h[1] = (_Float16)o.y;
    h[2] = (_Float16)o.z; h[3] = (_Float16)o.w;
    *(v4h*)(yhr + c * 4) = h;                 // RT: consumed by GEMM soon
  }
}

// ---------------------------------------------------------------------------
// Kernel 2: NF4 double-dequant. One thread per 64-element quant block.
// codes/qs are read exactly once -> NT loads so the 201 MB stream does not
// evict the L2-resident Wh we are producing.
// ---------------------------------------------------------------------------
__global__ __launch_bounds__(256) void dequant_kernel(const int* __restrict__ codes,
                                                      const int* __restrict__ qs,
                                                      const float* __restrict__ qf,
                                                      const float* __restrict__ mean,
                                                      _Float16* __restrict__ Wh) {
  __shared__ float cb[16];
  if (threadIdx.x < 16) cb[threadIdx.x] = NF4_CB[threadIdx.x];
  __syncthreads();

  const int blk = blockIdx.x * 256 + threadIdx.x;   // < N_BLOCKS
  const float scale =
      (float)__builtin_nontemporal_load(qs + blk) / qf[blk >> 8] + mean[0];

  const v4i* cp = (const v4i*)(codes + (size_t)blk * 64);
  v4h* wp = (v4h*)(Wh + (size_t)blk * 64);
#pragma unroll
  for (int i = 0; i < 16; ++i) {
    const v4i c = __builtin_nontemporal_load(cp + i);
    v4h h;
    h[0] = (_Float16)(cb[c.x & 15] * scale);
    h[1] = (_Float16)(cb[c.y & 15] * scale);
    h[2] = (_Float16)(cb[c.z & 15] * scale);
    h[3] = (_Float16)(cb[c.w & 15] * scale);
    wp[i] = h;                                // RT: keep Wh dirty in L2
  }
}

// ---------------------------------------------------------------------------
// Kernel 3: GEMM  C[M,N] = A[M,K] * W[N,K]^T   (f16 in, f32 acc/out)
// 256 threads = 8 waves arranged 4(M) x 2(N); workgroup tile 128 x 128.
// Each wave: 32(M) x 64(N) via 8 accumulators.
// K-loop steps 128 halves (4 WMMA sub-steps); one WGP prefetch pair per
// iteration, issued before any fragment register is live (avoids the
// WMMA->VALU 4-NOP hazard seen when prefetch address adds land between WMMAs).
// Per sub-step all 12 b128 loads are issued before the 8 WMMAs with distinct
// fragment registers -> staggered partial s_wait_loadcnt.
//
// Fragment addressing used here (wave32; contiguous K, not the ISA table order):
//   A (16x32, f16): lane m = lane&15 ; halves 0..15 -> K = kb + 16*(lane>>4)+0..15
//   B (32x16, f16): lane n = lane&15 ; halves 0..15 -> K = kb + 16*(lane>>4)+0..15
//   C/D (16x16, f32): VGPR r -> M = r + 8*(lane>>4), N = lane&15.
// ---------------------------------------------------------------------------
union FragH { v16h f; v8h h[2]; };

#define WMMA_F16(acc, a, b)                                                    \
  acc = __builtin_amdgcn_wmma_f32_16x16x32_f16(false, (a).f, false, (b).f,     \
                                               (short)0, acc, false, false)

#define GEMM_SUBSTEP(kk)                                                       \
  {                                                                            \
    FragH a0, a1, b0, b1, b2, b3;                                              \
    a0.h[0] = *(const v8h*)(Ar0 + (kk));                                       \
    a0.h[1] = *(const v8h*)(Ar0 + (kk) + 8);                                   \
    a1.h[0] = *(const v8h*)(Ar1 + (kk));                                       \
    a1.h[1] = *(const v8h*)(Ar1 + (kk) + 8);                                   \
    b0.h[0] = *(const v8h*)(B0 + (kk));                                        \
    b0.h[1] = *(const v8h*)(B0 + (kk) + 8);                                    \
    b1.h[0] = *(const v8h*)(B1 + (kk));                                        \
    b1.h[1] = *(const v8h*)(B1 + (kk) + 8);                                    \
    b2.h[0] = *(const v8h*)(B2 + (kk));                                        \
    b2.h[1] = *(const v8h*)(B2 + (kk) + 8);                                    \
    b3.h[0] = *(const v8h*)(B3 + (kk));                                        \
    b3.h[1] = *(const v8h*)(B3 + (kk) + 8);                                    \
    WMMA_F16(acc00, a0, b0);                                                   \
    WMMA_F16(acc10, a1, b0);                                                   \
    WMMA_F16(acc01, a0, b1);                                                   \
    WMMA_F16(acc11, a1, b1);                                                   \
    WMMA_F16(acc02, a0, b2);                                                   \
    WMMA_F16(acc12, a1, b2);                                                   \
    WMMA_F16(acc03, a0, b3);                                                   \
    WMMA_F16(acc13, a1, b3);                                                   \
  }

__global__ __launch_bounds__(256) void gemm_kernel(const _Float16* __restrict__ A,
                                                   const _Float16* __restrict__ W,
                                                   float* __restrict__ C) {
  const int K = IN_SIZE;
  const int N = OUT_SIZE;
  const int lane = threadIdx.x & 31;
  const int wave = threadIdx.x >> 5;
  const int l16 = lane >> 4;      // 0/1 half-wave
  const int lm  = lane & 15;

  const int m0 = blockIdx.y * 128 + (wave >> 1) * 32;
  const int n0 = blockIdx.x * 128 + (wave & 1) * 64;

  const _Float16* Ar0 = A + (size_t)(m0 + lm) * K + l16 * 16;
  const _Float16* Ar1 = Ar0 + (size_t)16 * K;
  const _Float16* B0 = W + (size_t)(n0 +  0 + lm) * K + l16 * 16;
  const _Float16* B1 = W + (size_t)(n0 + 16 + lm) * K + l16 * 16;
  const _Float16* B2 = W + (size_t)(n0 + 32 + lm) * K + l16 * 16;
  const _Float16* B3 = W + (size_t)(n0 + 48 + lm) * K + l16 * 16;

  v8f acc00 = {}, acc01 = {}, acc02 = {}, acc03 = {};
  v8f acc10 = {}, acc11 = {}, acc12 = {}, acc13 = {};

  for (int k = 0; k < K; k += 128) {
    // One WGP-scope prefetch pair per 256-byte A group, 2 groups ahead,
    // issued while no fragment registers are live.
    PREFETCH_WGP_512(Ar0 + k);
    PREFETCH_WGP_512(Ar1 + k);

    GEMM_SUBSTEP(k);
    GEMM_SUBSTEP(k + 32);
    GEMM_SUBSTEP(k + 64);
    GEMM_SUBSTEP(k + 96);
  }

  // NT stores: C is a 192 MB pure output stream; keep it out of L2.
  float* Cb0 = C + (size_t)(m0 + l16 * 8) * N + n0 + lm;
  float* Cb1 = Cb0 + (size_t)16 * N;
#pragma unroll
  for (int r = 0; r < 8; ++r) {
    const size_t ro = (size_t)r * N;
    __builtin_nontemporal_store(acc00[r], Cb0 + ro +  0);
    __builtin_nontemporal_store(acc01[r], Cb0 + ro + 16);
    __builtin_nontemporal_store(acc02[r], Cb0 + ro + 32);
    __builtin_nontemporal_store(acc03[r], Cb0 + ro + 48);
    __builtin_nontemporal_store(acc10[r], Cb1 + ro +  0);
    __builtin_nontemporal_store(acc11[r], Cb1 + ro + 16);
    __builtin_nontemporal_store(acc12[r], Cb1 + ro + 32);
    __builtin_nontemporal_store(acc13[r], Cb1 + ro + 48);
  }
}

// ---------------------------------------------------------------------------
// Launch
// ---------------------------------------------------------------------------
extern "C" void kernel_launch(void* const* d_in, const int* in_sizes, int n_in,
                              void* d_out, int out_size, void* d_ws, size_t ws_size,
                              hipStream_t stream) {
  const float* x     = (const float*)d_in[0];   // [4096, 4096]
  const float* gamma = (const float*)d_in[1];   // [4096]
  const float* beta  = (const float*)d_in[2];   // [4096]
  const float* qf    = (const float*)d_in[3];   // [3072]
  const float* mean  = (const float*)d_in[4];   // [1]
  const int*   codes = (const int*)d_in[5];     // [786432, 64]
  const int*   qs    = (const int*)d_in[6];     // [786432]

  float* lin_out = (float*)d_out;                                // [4096, 12288]
  float* ln_out  = lin_out + (size_t)M_ROWS * OUT_SIZE;          // [4096, 4096]

  _Float16* Ah = (_Float16*)d_ws;                                // 32 MB
  _Float16* Wh = Ah + (size_t)M_ROWS * IN_SIZE;                  // 96 MB (L2-resident)

  ln_kernel<<<M_ROWS, 256, 0, stream>>>(x, gamma, beta, ln_out, Ah);
  dequant_kernel<<<N_BLOCKS / 256, 256, 0, stream>>>(codes, qs, qf, mean, Wh);
  gemm_kernel<<<dim3(OUT_SIZE / 128, M_ROWS / 128), 256, 0, stream>>>(Ah, Wh, lin_out);
}
